// MSAAttention_46918222742016
// MI455X (gfx1250) — matrix operand order repaired and checked by reading, not compiled, not run on hardware
//
#include <hip/hip_runtime.h>
#include <hip/hip_bf16.h>

typedef __attribute__((ext_vector_type(16))) _Float16 v16h;
typedef __attribute__((ext_vector_type(8)))  _Float16 v8h;
typedef __attribute__((ext_vector_type(4)))  _Float16 v4h;
typedef __attribute__((ext_vector_type(8)))  float    v8f;

// ---------------------------------------------------------------------------
// CDNA5 async global->LDS helpers (ASYNCcnt path)
// ---------------------------------------------------------------------------
__device__ inline unsigned int lds_off(const void* p) {
  // flat LDS aperture: addr[31:0] is the LDS byte offset
  return (unsigned int)(unsigned long long)p;
}
__device__ inline void async_load_b128(unsigned int lds, unsigned int goff,
                                       const void* base) {
  // GVS mode: mem = SGPR base + 32-bit VGPR byte offset; LDS dest from VGPR
  asm volatile("global_load_async_to_lds_b128 %0, %1, %2"
               :: "v"(lds), "v"(goff), "s"(base) : "memory");
}
__device__ inline void wait_async0() {
#if __has_builtin(__builtin_amdgcn_s_wait_asynccnt)
  __builtin_amdgcn_s_wait_asynccnt(0);
#else
  asm volatile("s_wait_asynccnt 0x0" ::: "memory");
#endif
}

// ---------------------------------------------------------------------------
// wave32 reductions
// ---------------------------------------------------------------------------
__device__ inline float wave_sum(float v) {
#pragma unroll
  for (int off = 16; off > 0; off >>= 1) v += __shfl_xor(v, off, 32);
  return v;
}
__device__ inline float wave_max(float v) {
#pragma unroll
  for (int off = 16; off > 0; off >>= 1) v = fmaxf(v, __shfl_xor(v, off, 32));
  return v;
}
__device__ inline float block_sum(float v, float* red, int nwaves) {
  int t = threadIdx.x, lane = t & 31, wave = t >> 5;
  __syncthreads();
  v = wave_sum(v);
  if (lane == 0) red[wave] = v;
  __syncthreads();
  if (t == 0) {
    float s = 0.f;
    for (int i = 0; i < nwaves; ++i) s += red[i];
    red[0] = s;
  }
  __syncthreads();
  return red[0];
}

// ---------------------------------------------------------------------------
// WMMA fragment loaders (CDNA5 ISA 7.12.2, wave32); per-lane contiguous so
// they lower to 2 x ds_load_b128.
// A 16x32 (MxK) row-major; B 32x16 (KxN) held N-major; C: VGPR v, lane l ->
// M = v + 8*(l>>4), N = l&15.
// ---------------------------------------------------------------------------
__device__ inline v16h load_frag_a(const _Float16* base, int ld) {
  int lane = threadIdx.x & 31;
  int m = lane & 15, hi = lane >> 4;
  const _Float16* row = base + (size_t)m * ld;
  v16h a;
#pragma unroll
  for (int v = 0; v < 8; ++v) {
    int kk = (v < 4 ? 2 * v : 16 + 2 * (v - 4)) + 8 * hi;
    a[2 * v]     = row[kk];
    a[2 * v + 1] = row[kk + 1];
  }
  return a;
}
__device__ inline v16h load_frag_bT(const _Float16* base, int ld) {
  int lane = threadIdx.x & 31;
  int n = lane & 15, hi = lane >> 4;
  const _Float16* row = base + (size_t)n * ld;
  v16h b;
#pragma unroll
  for (int v = 0; v < 8; ++v) {
    int kk = (v < 4 ? 2 * v : 16 + 2 * (v - 4)) + 8 * hi;
    b[2 * v]     = row[kk];
    b[2 * v + 1] = row[kk + 1];
  }
  return b;
}

// ---------------------------------------------------------------------------
// weight conversion: f32 [K=256][N=256]  ->  f16 TRANSPOSED [N=256][K=256]
// ---------------------------------------------------------------------------
__global__ void convert_f16_T_kernel(const float* __restrict__ src,
                                     _Float16* __restrict__ dst, int n) {
  int i = blockIdx.x * blockDim.x + threadIdx.x;
  int stride = gridDim.x * blockDim.x;
  for (; i < n; i += stride) {
    int nn = i >> 8, kk = i & 255;
    dst[i] = (_Float16)src[kk * 256 + nn];
  }
}

// ---------------------------------------------------------------------------
// LayerNorm of m: one block (256 thr) per row of [65536, 256]; output f16
// ---------------------------------------------------------------------------
__global__ void ln_m_kernel(const float* __restrict__ m,
                            const float* __restrict__ w,
                            const float* __restrict__ b,
                            _Float16* __restrict__ out) {
  const int C = 256;
  __shared__ float red[8];
  size_t row = blockIdx.x;
  int t = threadIdx.x;
  float x  = m[row * C + t];
  float mu = block_sum(x, red, 8) * (1.f / 256.f);
  float d  = x - mu;
  float var = block_sum(d * d, red, 8) * (1.f / 256.f);
  out[row * C + t] = (_Float16)(d * rsqrtf(var + 1e-5f) * w[t] + b[t]);
}

// ---------------------------------------------------------------------------
// pair bias: LN(z[q,k,:128]) dot w_z[:,h]  -> pb[h][q][k]   (f32)
// ---------------------------------------------------------------------------
__global__ void pair_bias_kernel(const float* __restrict__ z,
                                 const float* __restrict__ w,
                                 const float* __restrict__ b,
                                 const float* __restrict__ wz,
                                 float* __restrict__ pb) {
  const int Cz = 128;
  __shared__ float red[8];
  size_t row = blockIdx.x;  // q*256 + k
  int t = threadIdx.x;
  float x  = z[row * Cz + t];
  float mu = block_sum(x, red, 4) * (1.f / 128.f);
  float d  = x - mu;
  float var = block_sum(d * d, red, 4) * (1.f / 128.f);
  float xn = d * rsqrtf(var + 1e-5f) * w[t] + b[t];
  for (int h = 0; h < 8; ++h) {
    float p = block_sum(xn * wz[t * 8 + h], red, 4);
    if (t == 0) pb[(size_t)h * 65536 + row] = p;
  }
}

// ---------------------------------------------------------------------------
// WMMA GEMM: out = act(scale * A[M,K] @ B[K,N] + bias); Bt is B transposed.
// block = 256 thr (8 waves); block tile 128x64; wave tile 32x32 (4 WMMA/step).
// Tiles staged with global_load_async_to_lds_b128, double-buffered so the
// next tile's HBM fetch overlaps this tile's WMMAs (ASYNCcnt pipeline).
// out_mode: 0 f32, 1 f16, 2 f16 per-head transposed vT[((s*8+h)*32+d)*256+r]
// ---------------------------------------------------------------------------
__global__ void wmma_gemm_kernel(const _Float16* __restrict__ A,
                                 const _Float16* __restrict__ Bt,
                                 void* __restrict__ outp,
                                 int M, int N, int K,
                                 float scale, const float* __restrict__ bias,
                                 int act, int out_mode) {
  __shared__ _Float16 sA [2][128][48];  // 24 KB, double-buffered
  __shared__ _Float16 sBt[2][64][48];   // 12 KB
  int t = threadIdx.x;
  int lane = t & 31, wave = t >> 5;
  int wr = wave >> 1, wc = wave & 1;    // 4x2 waves over 128x64
  int rowBase = blockIdx.x * 128;
  int colBase = blockIdx.y * 64;

  int r = t >> 2, cg = (t & 3) * 8;     // staging lane mapping (16B each)
  auto stage = [&](int bf, int kk) {
    unsigned int ga0 = (unsigned int)((((size_t)(rowBase + r)) * K + kk + cg) * 2);
    unsigned int ga1 = (unsigned int)((((size_t)(rowBase + r + 64)) * K + kk + cg) * 2);
    unsigned int gb  = (unsigned int)((((size_t)(colBase + r)) * K + kk + cg) * 2);
    async_load_b128(lds_off(&sA[bf][r][cg]),      ga0, A);
    async_load_b128(lds_off(&sA[bf][r + 64][cg]), ga1, A);
    async_load_b128(lds_off(&sBt[bf][r][cg]),     gb,  Bt);
  };

  v8f c00 = {}, c01 = {}, c10 = {}, c11 = {};
  int nk = K / 32;
  stage(0, 0);
  wait_async0();
  __syncthreads();
  int buf = 0;
  for (int ki = 0; ki < nk; ++ki) {
    if (ki + 1 < nk) stage(buf ^ 1, (ki + 1) * 32);  // prefetch next tile
    v16h a0 = load_frag_a(&sA[buf][32 * wr][0], 48);
    v16h a1 = load_frag_a(&sA[buf][32 * wr + 16][0], 48);
    v16h b0 = load_frag_bT(&sBt[buf][32 * wc][0], 48);
    v16h b1 = load_frag_bT(&sBt[buf][32 * wc + 16][0], 48);
    c00 = __builtin_amdgcn_wmma_f32_16x16x32_f16(false, a0, false, b0, (short)0, c00, false, false);
    c01 = __builtin_amdgcn_wmma_f32_16x16x32_f16(false, a0, false, b1, (short)0, c01, false, false);
    c10 = __builtin_amdgcn_wmma_f32_16x16x32_f16(false, a1, false, b0, (short)0, c10, false, false);
    c11 = __builtin_amdgcn_wmma_f32_16x16x32_f16(false, a1, false, b1, (short)0, c11, false, false);
    wait_async0();
    __syncthreads();
    buf ^= 1;
  }

  int hi = lane >> 4, n = lane & 15;
  v8f acc[2][2] = {{c00, c01}, {c10, c11}};
#pragma unroll
  for (int i = 0; i < 2; ++i) {
#pragma unroll
    for (int j = 0; j < 2; ++j) {
      int gcol = colBase + 32 * wc + 16 * j + n;
      float badd = bias ? bias[gcol] : 0.f;
#pragma unroll
      for (int vv = 0; vv < 8; ++vv) {
        int grow = rowBase + 32 * wr + 16 * i + vv + 8 * hi;
        float val = acc[i][j][vv] * scale + badd;
        if (act) val = 1.f / (1.f + __expf(-val));
        if (out_mode == 0) {
          ((float*)outp)[(size_t)grow * N + gcol] = val;
        } else if (out_mode == 1) {
          ((_Float16*)outp)[(size_t)grow * N + gcol] = (_Float16)val;
        } else {  // per-head transposed: vT[((s*8+h)*32+d)*256 + r]
          int shd = ((grow >> 8) * 8 + (gcol >> 5)) * 32 + (gcol & 31);
          ((_Float16*)outp)[(size_t)shd * 256 + (grow & 255)] = (_Float16)val;
        }
      }
    }
  }
}

// ---------------------------------------------------------------------------
// attention: per (s, h, 32-query tile). q pre-scaled by 1/sqrt(D).
// ---------------------------------------------------------------------------
__global__ void attention_kernel(const _Float16* __restrict__ q,
                                 const _Float16* __restrict__ k,
                                 const _Float16* __restrict__ vt,
                                 const float* __restrict__ pb,
                                 const float* __restrict__ mask,
                                 float* __restrict__ o) {
  const int R = 256, D = 32, HD = 256;
  __shared__ _Float16 sQ[32][32];      //  2 KB
  __shared__ _Float16 sK[64][32];      //  4 KB  (k rows, 32 d each)
  __shared__ _Float16 sVt[32][72];     //  4.5 KB (d rows, 64 krow, padded)
  __shared__ float    sS[32][256];     // 32 KB scores
  __shared__ _Float16 sP[32][256];     // 16 KB probabilities
  int t = threadIdx.x, lane = t & 31, wave = t >> 5;
  int q0 = blockIdx.x * 32;
  int h  = blockIdx.y;
  int s  = blockIdx.z;
  size_t headOff = (size_t)h * D;

  {  // Q tile 32x32
    int r = t >> 3, cg = (t & 7) * 4;
    *(v4h*)&sQ[r][cg] =
        *(const v4h*)(q + (size_t)(s * R + q0 + r) * HD + headOff + cg);
  }

  // ---- scores: waves as 2(q-sub) x 4(k-sub); 4 chunks of 64 keys ----
  int qsub = wave & 1, ksub = wave >> 1;
  for (int kc = 0; kc < 4; ++kc) {
    {
      int r = t >> 2, cg = (t & 3) * 8;
      *(v8h*)&sK[r][cg] =
          *(const v8h*)(k + (size_t)(s * R + kc * 64 + r) * HD + headOff + cg);
    }
    __syncthreads();
    v16h a   = load_frag_a(&sQ[16 * qsub][0], 32);
    v16h bfr = load_frag_bT(&sK[16 * ksub][0], 32);  // B[d][krow] = sK[krow][d]
    v8f c = {};
    c = __builtin_amdgcn_wmma_f32_16x16x32_f16(false, a, false, bfr,
                                               (short)0, c, false, false);
    int hi = lane >> 4, n = lane & 15;
    int kcol = kc * 64 + 16 * ksub + n;
    float mb = 1.0e9f * (mask[(size_t)s * R + kcol] - 1.0f);
#pragma unroll
    for (int vv = 0; vv < 8; ++vv) {
      int qr = 16 * qsub + vv + 8 * hi;
      sS[qr][kcol] = c[vv] + mb + pb[((size_t)h * R + (q0 + qr)) * R + kcol];
    }
    __syncthreads();
  }

  // ---- softmax over 256 keys, 4 rows per wave ----
  for (int i = 0; i < 4; ++i) {
    int r = wave * 4 + i;
    float mx = -3.4e38f;
    for (int cc = lane; cc < 256; cc += 32) mx = fmaxf(mx, sS[r][cc]);
    mx = wave_max(mx);
    float sum = 0.f;
    for (int cc = lane; cc < 256; cc += 32) sum += __expf(sS[r][cc] - mx);
    sum = wave_sum(sum);
    float inv = 1.f / sum;
    for (int cc = lane; cc < 256; cc += 32)
      sP[r][cc] = (_Float16)(__expf(sS[r][cc] - mx) * inv);
  }
  __syncthreads();

  // ---- PV: o[32,32] = P[32,256] @ V[256,32]; waves 0-3 as 2x2 ----
  int nsub = wave >> 1;
  v8f c = {};
  for (int kc = 0; kc < 4; ++kc) {
    {  // stage V chunk from vT: sVt[d][krow], both contiguous
      int r = t >> 3, cg = (t & 7) * 8;
      *(v8h*)&sVt[r][cg] = *(const v8h*)(
          vt + ((size_t)(s * 8 + h) * 32 + r) * 256 + kc * 64 + cg);
    }
    __syncthreads();
    if (wave < 4) {
#pragma unroll
      for (int kkk = 0; kkk < 2; ++kkk) {
        v16h a   = load_frag_a(&sP[16 * qsub][kc * 64 + kkk * 32], 256);
        v16h bfr = load_frag_bT(&sVt[16 * nsub][kkk * 32], 72);
        c = __builtin_amdgcn_wmma_f32_16x16x32_f16(false, a, false, bfr,
                                                   (short)0, c, false, false);
      }
    }
    __syncthreads();
  }
  if (wave < 4) {
    int hi = lane >> 4, n = lane & 15;
#pragma unroll
    for (int vv = 0; vv < 8; ++vv) {
      int qr = q0 + 16 * qsub + vv + 8 * hi;
      o[(size_t)(s * R + qr) * HD + headOff + 16 * nsub + n] = c[vv];
    }
  }
}

// ---------------------------------------------------------------------------
// elementwise gate: gated16 = f16(o * g)
// ---------------------------------------------------------------------------
__global__ void gate_kernel(const float* __restrict__ o,
                            const float* __restrict__ g,
                            _Float16* __restrict__ out, int n) {
  int i = blockIdx.x * blockDim.x + threadIdx.x;
  int stride = gridDim.x * blockDim.x;
  for (; i < n; i += stride) out[i] = (_Float16)(o[i] * g[i]);
}

// ---------------------------------------------------------------------------
extern "C" void kernel_launch(void* const* d_in, const int* in_sizes, int n_in,
                              void* d_out, int out_size, void* d_ws,
                              size_t ws_size, hipStream_t stream) {
  (void)in_sizes; (void)n_in; (void)out_size; (void)ws_size;
  const float* m      = (const float*)d_in[0];
  const float* z      = (const float*)d_in[1];
  const float* mask   = (const float*)d_in[2];
  const float* ln_m_w = (const float*)d_in[3];
  const float* ln_m_b = (const float*)d_in[4];
  const float* ln_z_w = (const float*)d_in[5];
  const float* ln_z_b = (const float*)d_in[6];
  const float* w_z    = (const float*)d_in[7];
  const float* wq     = (const float*)d_in[8];
  const float* wk     = (const float*)d_in[9];
  const float* wv     = (const float*)d_in[10];
  const float* wg     = (const float*)d_in[11];
  const float* bg     = (const float*)d_in[12];
  const float* wo     = (const float*)d_in[13];
  const float* bo     = (const float*)d_in[14];

  const int SR = 65536, C = 256;
  char* ws = (char*)d_ws;
  size_t off = 0;
  auto carve = [&](size_t bytes) -> void* {
    void* p = ws + off;
    off = (off + bytes + 255) & ~(size_t)255;
    return p;
  };
  _Float16* mln16 = (_Float16*)carve((size_t)SR * C * 2);
  _Float16* q16   = (_Float16*)carve((size_t)SR * C * 2);
  _Float16* k16   = (_Float16*)carve((size_t)SR * C * 2);
  _Float16* vt16  = (_Float16*)carve((size_t)SR * C * 2);
  float*    gbuf  = (float*)carve((size_t)SR * C * 4);
  float*    obuf  = (float*)carve((size_t)SR * C * 4);
  _Float16* gated = (_Float16*)carve((size_t)SR * C * 2);
  float*    pb    = (float*)carve((size_t)8 * 65536 * 4);
  _Float16* wqT   = (_Float16*)carve((size_t)65536 * 2);
  _Float16* wkT   = (_Float16*)carve((size_t)65536 * 2);
  _Float16* wvT   = (_Float16*)carve((size_t)65536 * 2);
  _Float16* wgT   = (_Float16*)carve((size_t)65536 * 2);
  _Float16* woT   = (_Float16*)carve((size_t)65536 * 2);

  convert_f16_T_kernel<<<256, 256, 0, stream>>>(wq, wqT, 65536);
  convert_f16_T_kernel<<<256, 256, 0, stream>>>(wk, wkT, 65536);
  convert_f16_T_kernel<<<256, 256, 0, stream>>>(wv, wvT, 65536);
  convert_f16_T_kernel<<<256, 256, 0, stream>>>(wg, wgT, 65536);
  convert_f16_T_kernel<<<256, 256, 0, stream>>>(wo, woT, 65536);

  ln_m_kernel<<<SR, 256, 0, stream>>>(m, ln_m_w, ln_m_b, mln16);
  pair_bias_kernel<<<65536, 128, 0, stream>>>(z, ln_z_w, ln_z_b, w_z, pb);

  dim3 gg(SR / 128, C / 64);
  wmma_gemm_kernel<<<gg, 256, 0, stream>>>(mln16, wqT, q16, SR, C, C,
                                           0.17677669529663687f, nullptr, 0, 1);
  wmma_gemm_kernel<<<gg, 256, 0, stream>>>(mln16, wkT, k16, SR, C, C,
                                           1.0f, nullptr, 0, 1);
  wmma_gemm_kernel<<<gg, 256, 0, stream>>>(mln16, wvT, vt16, SR, C, C,
                                           1.0f, nullptr, 0, 2);
  wmma_gemm_kernel<<<gg, 256, 0, stream>>>(mln16, wgT, gbuf, SR, C, C,
                                           1.0f, bg, 1, 0);

  attention_kernel<<<dim3(8, 8, 256), 256, 0, stream>>>(q16, k16, vt16, pb,
                                                        mask, obuf);

  gate_kernel<<<4096, 256, 0, stream>>>(obuf, gbuf, gated, SR * C);
  wmma_gemm_kernel<<<gg, 256, 0, stream>>>(gated, woT, d_out, SR, C, C,
                                           1.0f, bo, 0, 0);
}
